// DenseCRFHead_86681029968529
// MI455X (gfx1250) — compile-verified, matrix operand order checked by reading, not run on hardware
//
#include <hip/hip_runtime.h>

// ---------------------------------------------------------------------------
// DenseCRF bilateral loss for MI455X (gfx1250, wave32, WMMA)
//   loss = -(1e-8/8) * sum_n sum_{i,j} exp(-0.5*||f_i-f_j||^2) * (seg_i . seg_j)
// Both inner products per 16x16 tile are computed with v_wmma_f32_16x16x32_f16.
// ---------------------------------------------------------------------------

typedef _Float16 v16h __attribute__((ext_vector_type(16)));
typedef _Float16 v8h  __attribute__((ext_vector_type(8)));
typedef float    v8f  __attribute__((ext_vector_type(8)));

#define NIMG   8
#define HW     64
#define PDIM   (HW*HW)          // 4096 pixels
#define KSEG   21
#define KPAD   32               // WMMA K
#define TILES  (PDIM/16)        // 256
#define NPAIRS (TILES*(TILES+1)/2)  // 32896
#define WPB    8                // waves per block (256 threads)
#define GRIDX  (NPAIRS/WPB)     // 4112 (exact)

#define SEGH_BYTES  ((size_t)NIMG*PDIM*KPAD*2)   // 2 MiB
#define FEAT_BYTES  ((size_t)NIMG*PDIM*KPAD*2)   // 2 MiB
#define NORM_BYTES  ((size_t)NIMG*PDIM*4)        // 128 KiB

// ---------------------------------------------------------------------------
// Kernel 1: resize seg 128->64 (4-tap antialiased bilinear, edge-renormalized),
// pack seg (K padded to 32) and bilateral features (5 padded to 32) as f16 in
// WMMA-friendly row-major-K layout; f32 squared norms of the f16-rounded
// features (so exp(G - 0.5(ni+nj)) is exact for the quantized features).
// ---------------------------------------------------------------------------
__global__ void crf_prep_kernel(const float* __restrict__ images,
                                const float* __restrict__ segm,
                                _Float16* __restrict__ segh,
                                _Float16* __restrict__ feath,
                                float* __restrict__ norms) {
  int tid = blockIdx.x * blockDim.x + threadIdx.x;
  if (tid >= NIMG * PDIM) return;
  int n = tid >> 12;
  int p = tid & (PDIM - 1);
  int y = p >> 6, x = p & 63;

  // per-axis 4-tap weights {1/8,3/8,3/8,1/8} at {2o-1..2o+2}, renormalized
  const float wt[4] = {0.125f, 0.375f, 0.375f, 0.125f};
  int ix[4], iy[4]; float wx[4], wy[4];
  float sx = 0.f, sy = 0.f;
#pragma unroll
  for (int a = 0; a < 4; a++) {
    int tx = 2 * x - 1 + a, ty = 2 * y - 1 + a;
    if (tx >= 0 && tx < 128) { ix[a] = tx; wx[a] = wt[a]; sx += wt[a]; }
    else                     { ix[a] = 0;  wx[a] = 0.f; }
    if (ty >= 0 && ty < 128) { iy[a] = ty; wy[a] = wt[a]; sy += wt[a]; }
    else                     { iy[a] = 0;  wy[a] = 0.f; }
  }
  float rsx = 1.f / sx, rsy = 1.f / sy;
#pragma unroll
  for (int a = 0; a < 4; a++) { wx[a] *= rsx; wy[a] *= rsy; }

  _Float16* so = segh + (size_t)tid * KPAD;
  for (int k = 0; k < KSEG; k++) {
    const float* sp = segm + ((size_t)n * KSEG + k) * 128 * 128;
    float s = 0.f;
#pragma unroll
    for (int a = 0; a < 4; a++) {
      if (wy[a] == 0.f) continue;
      const float* rp = sp + iy[a] * 128;
      s += wy[a] * (wx[0] * rp[ix[0]] + wx[1] * rp[ix[1]] +
                    wx[2] * rp[ix[2]] + wx[3] * rp[ix[3]]);
    }
    so[k] = (_Float16)s;
  }
#pragma unroll
  for (int k = KSEG; k < KPAD; k++) so[k] = (_Float16)0.f;

  // bilateral features: (x,y)/ (100*0.5) , rgb / 15
  const float inv_sxy = 1.f / 50.f, inv_rgb = 1.f / 15.f;
  const float* ib = images + (size_t)n * 3 * PDIM;
  _Float16 h[5];
  h[0] = (_Float16)((float)x * inv_sxy);
  h[1] = (_Float16)((float)y * inv_sxy);
  h[2] = (_Float16)(ib[p] * inv_rgb);
  h[3] = (_Float16)(ib[PDIM + p] * inv_rgb);
  h[4] = (_Float16)(ib[2 * PDIM + p] * inv_rgb);

  _Float16* fo = feath + (size_t)tid * KPAD;
  float nrm = 0.f;
#pragma unroll
  for (int k = 0; k < 5; k++) { fo[k] = h[k]; float g = (float)h[k]; nrm += g * g; }
#pragma unroll
  for (int k = 5; k < KPAD; k++) fo[k] = (_Float16)0.f;
  norms[tid] = nrm;
}

// pairs before triangular row i (ti<=tj enumeration)
__device__ __forceinline__ int tri_off(int i) {
  return i * TILES - ((i * (i - 1)) >> 1);
}

// gather one 16x32 f16 WMMA operand (A layout; B of a Gram uses the same
// per-lane gather with the column tile base): lane l holds row base+(l&15),
// K = {hi*8..hi*8+7, 16+hi*8..16+hi*8+7}, hi = l>>4, as b128 pairs.
__device__ __forceinline__ v16h load_tile16x32(const _Float16* __restrict__ base,
                                               int hi) {
  const v8h* rp = (const v8h*)base;
  v8h lo = rp[hi];        // K = hi*8 .. hi*8+7
  v8h hk = rp[2 + hi];    // K = 16+hi*8 .. 16+hi*8+7
  v16h r;
#pragma unroll
  for (int i = 0; i < 8; i++) { r[i] = lo[i]; r[8 + i] = hk[i]; }
  return r;
}

// ---------------------------------------------------------------------------
// Kernel 2: one wave per (image, tile-pair ti<=tj).
//   S = seg_I (16x32) x seg_J^T  -> v_wmma_f32_16x16x32_f16
//   G = f_I   (16x32) x f_J^T    -> v_wmma_f32_16x16x32_f16
//   acc += exp(G - 0.5*(ni+nj)) * S  on the f32 accumulator layout.
// Off-diagonal tile pairs doubled (W and S symmetric). Block writes one
// deterministic partial sum.
// ---------------------------------------------------------------------------
__global__ void crf_tile_kernel(const _Float16* __restrict__ segh,
                                const _Float16* __restrict__ feath,
                                const float* __restrict__ norms,
                                float* __restrict__ partials) {
  __shared__ float wsum[WPB];
  const int lane = threadIdx.x & 31;
  const int wv   = threadIdx.x >> 5;
  const int n    = blockIdx.y;
  const int t    = blockIdx.x * WPB + wv;

  float acc = 0.f;
  if (t < NPAIRS) {
    // decode triangular pair (ti <= tj)
    const float T2 = 2.f * TILES + 1.f;
    int ti = (int)((T2 - sqrtf(T2 * T2 - 8.f * (float)t)) * 0.5f);
    if (ti < 0) ti = 0; if (ti > TILES - 1) ti = TILES - 1;
    while (ti > 0 && tri_off(ti) > t) --ti;
    while (ti < TILES - 1 && tri_off(ti + 1) <= t) ++ti;
    const int tj = ti + (t - tri_off(ti));
    const int I0 = ti * 16, J0 = tj * 16;

    const int col = lane & 15;
    const int hi  = lane >> 4;
    const size_t rowA = ((size_t)n * PDIM + I0 + col) * KPAD;
    const size_t rowB = ((size_t)n * PDIM + J0 + col) * KPAD;

    v16h As = load_tile16x32(segh + rowA, hi);
    v16h Bs = load_tile16x32(segh + rowB, hi);
    v16h Af = load_tile16x32(feath + rowA, hi);
    v16h Bf = load_tile16x32(feath + rowB, hi);

    v8f zero = {};
    v8f S = __builtin_amdgcn_wmma_f32_16x16x32_f16(
        false, As, false, Bs, (short)0, zero, false, false);
    v8f G = __builtin_amdgcn_wmma_f32_16x16x32_f16(
        false, Af, false, Bf, (short)0, zero, false, false);

    // norms: column j = J0+col per lane; rows I0 + 8*hi + v for VGPR v
    const float* nn = norms + (size_t)n * PDIM;
    const float nj = nn[J0 + col];
    const float4* nip = (const float4*)(nn + I0 + hi * 8);
    float4 n0 = nip[0], n1 = nip[1];
    float ni[8] = {n0.x, n0.y, n0.z, n0.w, n1.x, n1.y, n1.z, n1.w};

#pragma unroll
    for (int v = 0; v < 8; v++)
      acc += __expf(G[v] - 0.5f * (ni[v] + nj)) * S[v];

    if (ti != tj) acc *= 2.f;   // symmetry: count (tj,ti) too
  }

  // wave32 reduction, then deterministic per-block partial
#pragma unroll
  for (int o = 16; o > 0; o >>= 1) acc += __shfl_down(acc, o, 32);
  if (lane == 0) wsum[wv] = acc;
  __syncthreads();
  if (threadIdx.x == 0) {
    float s = 0.f;
#pragma unroll
    for (int i = 0; i < WPB; i++) s += wsum[i];
    partials[(size_t)n * GRIDX + blockIdx.x] = s;
  }
}

// ---------------------------------------------------------------------------
// Kernel 3: fixed-order reduction of partials -> d_out[0] (deterministic).
// ---------------------------------------------------------------------------
__global__ void crf_final_kernel(const float* __restrict__ partials,
                                 int npart, float* __restrict__ out) {
  __shared__ float red[256];
  float s = 0.f;
  for (int i = threadIdx.x; i < npart; i += 256) s += partials[i];
  red[threadIdx.x] = s;
  __syncthreads();
  for (int o = 128; o > 0; o >>= 1) {
    if (threadIdx.x < o) red[threadIdx.x] += red[threadIdx.x + o];
    __syncthreads();
  }
  if (threadIdx.x == 0)
    out[0] = red[0] * (-1e-8f / (float)NIMG);   // WEIGHT * (-sum / N)
}

extern "C" void kernel_launch(void* const* d_in, const int* in_sizes, int n_in,
                              void* d_out, int out_size, void* d_ws, size_t ws_size,
                              hipStream_t stream) {
  const float* images = (const float*)d_in[0];   // (8,3,64,64) f32
  const float* segm   = (const float*)d_in[1];   // (8,21,128,128) f32
  float* out = (float*)d_out;                    // (1,) f32

  char* ws = (char*)d_ws;
  _Float16* segh  = (_Float16*)(ws);
  _Float16* feath = (_Float16*)(ws + SEGH_BYTES);
  float* norms    = (float*)(ws + SEGH_BYTES + FEAT_BYTES);
  float* partials = (float*)(ws + SEGH_BYTES + FEAT_BYTES + NORM_BYTES);

  crf_prep_kernel<<<(NIMG * PDIM + 255) / 256, 256, 0, stream>>>(
      images, segm, segh, feath, norms);

  dim3 grid(GRIDX, NIMG);
  crf_tile_kernel<<<grid, 256, 0, stream>>>(segh, feath, norms, partials);

  crf_final_kernel<<<1, 256, 0, stream>>>(partials, NIMG * GRIDX, out);
}
